// AttentionBlock_52862457479271
// MI455X (gfx1250) — matrix operand order repaired and checked by reading, not compiled, not run on hardware
//
#include <hip/hip_runtime.h>
#include <hip/hip_bf16.h>

// ---------------------------------------------------------------------------
// AttentionBlock for MI455X (gfx1250): GroupNorm -> QKV -> flash attention
// (online softmax, bf16 WMMA f32-accum, async-to-LDS K/V staging) -> proj.
// Wave32, V_WMMA_F32_16X16X32_BF16 throughout.
// ---------------------------------------------------------------------------

typedef __attribute__((ext_vector_type(16))) __bf16 v16bf;
typedef __attribute__((ext_vector_type(8)))  float  v8f;
typedef __attribute__((ext_vector_type(4)))  int    v4i;

#define BD     2
#define CD     256
#define ND     4096      // 16*16*16
#define NGRP   8
#define EPSV   1e-5f

union Frag16 { v16bf v; uint4 q[2]; };

// A-fragment (16x32 bf16, MxK): lane holds row m=lane&15, two contiguous
// 8-element (16B) K-chunks at K = kh*16 and kh*16+8 (kh = lane>>4).
__device__ __forceinline__ v16bf load_frag_2x16B(const __bf16* p0, const __bf16* p1) {
  Frag16 f;
  f.q[0] = *reinterpret_cast<const uint4*>(p0);
  f.q[1] = *reinterpret_cast<const uint4*>(p1);
  return f.v;
}
// B-fragment (32x16 bf16, KxN): lane holds column n=lane&15, 16 contiguous
// K values starting at K = kh*16  -> one 32B contiguous load.
__device__ __forceinline__ v16bf load_frag_32B(const __bf16* p) {
  Frag16 f;
  const uint4* q = reinterpret_cast<const uint4*>(p);
  f.q[0] = q[0];
  f.q[1] = q[1];
  return f.v;
}
__device__ __forceinline__ v8f v8f_zero() {
  v8f z = {0.f, 0.f, 0.f, 0.f, 0.f, 0.f, 0.f, 0.f};
  return z;
}

// ---- CDNA5 async global->LDS copy (16B per lane), with safe fallback ------
// Probe round 2 confirmed the builtin exists with signature
// (AS1 v4i* src, AS3 v4i* dst, imm offset, imm cpol).
__device__ __forceinline__ void async_copy16(void* ldst, const void* gsrc) {
#if __has_builtin(__builtin_amdgcn_global_load_async_to_lds_b128)
  __builtin_amdgcn_global_load_async_to_lds_b128(
      (__attribute__((address_space(1))) v4i*)(gsrc),
      (__attribute__((address_space(3))) v4i*)(ldst),
      0, 0);
#else
  *reinterpret_cast<uint4*>(ldst) = *reinterpret_cast<const uint4*>(gsrc);
#endif
}
__device__ __forceinline__ void wait_async_le8() {
#if __has_builtin(__builtin_amdgcn_s_wait_asynccnt)
  __builtin_amdgcn_s_wait_asynccnt(8);
#else
  asm volatile("s_wait_asynccnt 0x8" ::: "memory");
#endif
}
__device__ __forceinline__ void wait_async_0() {
#if __has_builtin(__builtin_amdgcn_s_wait_asynccnt)
  __builtin_amdgcn_s_wait_asynccnt(0);
#else
  asm volatile("s_wait_asynccnt 0x0" ::: "memory");
#endif
}

// ---------------------------------------------------------------------------
// Kernel 1: GroupNorm over (b,g); writes bf16 xn transposed to [B][N][C]
// (C contiguous -> WMMA B-fragment friendly for the QKV GEMM).
// ---------------------------------------------------------------------------
__global__ void __launch_bounds__(256) gn_kernel(const float* __restrict__ x,
                                                 const float* __restrict__ w,
                                                 const float* __restrict__ bias,
                                                 __bf16* __restrict__ xnT) {
  const int bg = blockIdx.x;              // 0..15
  const int b = bg / NGRP, g = bg % NGRP;
  const int CPG = CD / NGRP;              // 32
  const int M = CPG * ND;                 // 131072 (contiguous in x)
  const float* base = x + (size_t)(b * CD + g * CPG) * ND;

  float s = 0.f, s2 = 0.f;
  for (int i = threadIdx.x; i < M; i += blockDim.x) {
    float v = base[i];
    s += v; s2 += v * v;
  }
  __shared__ float r1[256], r2[256];
  r1[threadIdx.x] = s; r2[threadIdx.x] = s2;
  __syncthreads();
  for (int off = 128; off > 0; off >>= 1) {
    if ((int)threadIdx.x < off) {
      r1[threadIdx.x] += r1[threadIdx.x + off];
      r2[threadIdx.x] += r2[threadIdx.x + off];
    }
    __syncthreads();
  }
  const float mean = r1[0] / (float)M;
  const float var  = r2[0] / (float)M - mean * mean;
  const float rstd = rsqrtf(var + EPSV);

  for (int i = threadIdx.x; i < M; i += blockDim.x) {
    int cl = i >> 12;               // i / 4096
    int n  = i & (ND - 1);
    int c  = g * CPG + cl;
    float v = (base[i] - mean) * rstd * w[c] + bias[c];
    xnT[((size_t)b * ND + n) * CD + c] = (__bf16)v;
  }
}

// ---------------------------------------------------------------------------
// Kernel 2: convert weights f32 -> bf16 (row-major [O][C] is already the
// WMMA A-fragment layout: contiguous along K=C).
// ---------------------------------------------------------------------------
__global__ void cvt_kernel(const float* __restrict__ qkv_w,
                           const float* __restrict__ proj_w,
                           __bf16* __restrict__ qw, __bf16* __restrict__ pw) {
  int i = blockIdx.x * blockDim.x + threadIdx.x;
  if (i < 3 * CD * CD) qw[i] = (__bf16)qkv_w[i];
  if (i < CD * CD)     pw[i] = (__bf16)proj_w[i];
}

// ---------------------------------------------------------------------------
// Kernel 3: QKV GEMM  qkv[o,n] = sum_c W[o,c] * xn[c,n]  via WMMA bf16.
// Block = 4 waves, 64(o) x 64(n) tile; wave = 16(o) x 64(n).
// Inner K-loop is software-pipelined one step ahead in registers.
// Stores: Q,K -> [B][N][C] (contig C), V -> [B][C][N] (contig N).
// ---------------------------------------------------------------------------
__global__ void __launch_bounds__(128) qkv_kernel(const __bf16* __restrict__ qw,
                                                  const float* __restrict__ qb,
                                                  const __bf16* __restrict__ xnT,
                                                  __bf16* __restrict__ Qb,
                                                  __bf16* __restrict__ Kb,
                                                  __bf16* __restrict__ Vb) {
  const int b    = blockIdx.z;
  const int o0   = blockIdx.x * 64;
  const int n0   = blockIdx.y * 64;
  const int wv   = threadIdx.x >> 5;
  const int lane = threadIdx.x & 31;
  const int col  = lane & 15;
  const int kh   = lane >> 4;
  const int orow = o0 + wv * 16;

  v8f acc[4];
#pragma unroll
  for (int t = 0; t < 4; ++t) acc[t] = v8f_zero();

  const __bf16* arow  = qw + (size_t)(orow + col) * CD;
  const __bf16* bbase = xnT + ((size_t)b * ND + n0) * CD;

  // prime the register pipeline (kc = 0)
  v16bf af = load_frag_2x16B(arow + kh * 16, arow + kh * 16 + 8);
  v16bf bf[4];
#pragma unroll
  for (int t = 0; t < 4; ++t)
    bf[t] = load_frag_32B(bbase + (size_t)(t * 16 + col) * CD + kh * 16);

#pragma unroll
  for (int kc = 0; kc < 8; ++kc) {
    v16bf afn, bfn[4];
    const bool more = (kc < 7);
    if (more) {                                  // issue next-K loads early
      const int c1 = (kc + 1) * 32;
      afn = load_frag_2x16B(arow + c1 + kh * 16, arow + c1 + kh * 16 + 8);
#pragma unroll
      for (int t = 0; t < 4; ++t)
        bfn[t] = load_frag_32B(bbase + (size_t)(t * 16 + col) * CD + c1 + kh * 16);
    }
#pragma unroll
    for (int t = 0; t < 4; ++t)
      acc[t] = __builtin_amdgcn_wmma_f32_16x16x32_bf16(false, af, false, bf[t],
                                                       (short)0, acc[t], false, false);
    if (more) {
      af = afn;
#pragma unroll
      for (int t = 0; t < 4; ++t) bf[t] = bfn[t];
    }
  }
#pragma unroll
  for (int t = 0; t < 4; ++t) {
#pragma unroll
    for (int j = 0; j < 8; ++j) {
      const int o = orow + j + 8 * kh;          // C/D layout: row = j + 8*(lane>=16)
      const int n = n0 + t * 16 + col;
      const float v = acc[t][j] + qb[o];
      if (o < CD)          Qb[((size_t)b * ND + n) * CD + o] = (__bf16)v;
      else if (o < 2 * CD) Kb[((size_t)b * ND + n) * CD + (o - CD)] = (__bf16)v;
      else                 Vb[((size_t)b * CD + (o - 2 * CD)) * ND + n] = (__bf16)v;
    }
  }
}

// ---------------------------------------------------------------------------
// Kernel 4: flash attention, head dim 256, online softmax.
// Block = 4 waves; each wave owns a 16-query tile, keys stepped by 32 so the
// probability tile P[16x32] is exactly one bf16 A-fragment.
//  - K tiles (32x256, 16KB) double-buffered in LDS, async-prefetched one
//    iteration ahead.
//  - V tiles (256x32, 16KB) single-buffered, async-issued at iteration top,
//    latency hidden under the QK^T WMMA phase.
//  - P is re-laid-out C/D->A through a per-wave LDS slab (intra-wave DS is
//    in-order; explicit s_wait_dscnt before the reload).
// LDS: 32KB (sK) + 16KB (sV) + 4KB (sP) = 52KB.
// ---------------------------------------------------------------------------
__global__ void __launch_bounds__(128) flash_kernel(const __bf16* __restrict__ Qb,
                                                    const __bf16* __restrict__ Kb,
                                                    const __bf16* __restrict__ Vb,
                                                    __bf16* __restrict__ Ob) {
  const int b    = blockIdx.y;
  const int q0   = blockIdx.x * 64;
  const int tid  = threadIdx.x;
  const int wv   = tid >> 5;
  const int lane = tid & 31;
  const int col  = lane & 15;
  const int kh   = lane >> 4;
  const int qrow = q0 + wv * 16;

  __shared__ __bf16 sK[2][32][CD];              // 2 x 16 KB, double buffered
  __shared__ __bf16 sV[CD][32];                 // 16 KB
  __shared__ __bf16 sP[4][16][32];              // 4 KB, per-wave P slab

  const __bf16* qbase = Qb + ((size_t)b * ND + qrow) * CD;
  const __bf16* kbase = Kb + (size_t)b * ND * CD;
  const __bf16* vbase = Vb + (size_t)b * CD * ND;

  // --- async tile copy helpers (1024 x 16B chunks, 8 per thread) -----------
  auto issueK = [&](int it, int pb) {
    const int nk0 = it * 32;
#pragma unroll
    for (int i = 0; i < 8; ++i) {
      const int idx = tid + 128 * i;            // 0..1023
      const int r   = idx >> 5;                 // key row 0..31 (512B/row)
      const int cc  = (idx & 31) * 8;           // bf16 elem offset in row
      async_copy16(&sK[pb][r][cc], kbase + (size_t)(nk0 + r) * CD + cc);
    }
  };
  auto issueV = [&](int it) {
    const int nk0 = it * 32;
#pragma unroll
    for (int i = 0; i < 8; ++i) {
      const int idx  = tid + 128 * i;           // 0..1023
      const int c    = idx >> 2;                // channel row 0..255 (64B/row)
      const int part = (idx & 3) * 8;
      async_copy16(&sV[c][part], vbase + (size_t)c * ND + nk0 + part);
    }
  };

  v8f oacc[16];                                 // 16 x (16q x 16c) f32 tiles
#pragma unroll
  for (int t = 0; t < 16; ++t) oacc[t] = v8f_zero();
  float mrow[8], lrow[8];
#pragma unroll
  for (int j = 0; j < 8; ++j) { mrow[j] = -3.0e38f; lrow[j] = 0.f; }

  const float scale = 0.0625f;                  // 1/sqrt(256)
  const __bf16* qr = qbase + (size_t)col * CD;  // this lane's Q row (A layout)

  const int NIT = ND / 32;                      // 128 key tiles

  // prologue: stage K tile 0
  issueK(0, 0);
  wait_async_0();
  __syncthreads();

  for (int it = 0; it < NIT; ++it) {
    const int p = it & 1;
    const bool pk = (it + 1 < NIT);

    issueV(it);                                 // V for this iteration
    if (pk) issueK(it + 1, p ^ 1);              // K prefetch, next iteration

    // ---- S = (Q K^T) * scale : 16 x 32, two 16-wide subtiles (from sK[p]) -
    v8f s0 = v8f_zero(), s1 = v8f_zero();
#pragma unroll
    for (int kc = 0; kc < 8; ++kc) {
      const int c0 = kc * 32;
      v16bf af = load_frag_2x16B(qr + c0 + kh * 16, qr + c0 + kh * 16 + 8);
      v16bf b0 = load_frag_32B(&sK[p][col][c0 + kh * 16]);
      s0 = __builtin_amdgcn_wmma_f32_16x16x32_bf16(false, af, false, b0,
                                                   (short)0, s0, false, false);
      v16bf b1 = load_frag_32B(&sK[p][16 + col][c0 + kh * 16]);
      s1 = __builtin_amdgcn_wmma_f32_16x16x32_bf16(false, af, false, b1,
                                                   (short)0, s1, false, false);
    }
    // ---- online softmax over these 32 keys --------------------------------
#pragma unroll
    for (int j = 0; j < 8; ++j) {
      float a0 = s0[j] * scale;
      float a1 = s1[j] * scale;
      float t = fmaxf(a0, a1);
#pragma unroll
      for (int msk = 8; msk >= 1; msk >>= 1)    // reduce within 16-lane halves
        t = fmaxf(t, __shfl_xor(t, msk, 32));
      const float newm = fmaxf(mrow[j], t);
      const float corr = __expf(mrow[j] - newm);
      const float p0 = __expf(a0 - newm);
      const float p1 = __expf(a1 - newm);
      float rs = p0 + p1;
#pragma unroll
      for (int msk = 8; msk >= 1; msk >>= 1)
        rs += __shfl_xor(rs, msk, 32);
      lrow[j] = lrow[j] * corr + rs;
      mrow[j] = newm;
#pragma unroll
      for (int t2 = 0; t2 < 16; ++t2) oacc[t2][j] *= corr;   // rescale row j
      // C/D layout -> row-major bf16 in LDS
      const int r = j + 8 * kh;
      sP[wv][r][col]      = (__bf16)p0;
      sP[wv][r][col + 16] = (__bf16)p1;
    }
    asm volatile("s_wait_dscnt 0x0" ::: "memory");
    // ---- reload P as one A-fragment (16x32) -------------------------------
    const __bf16* pr = &sP[wv][col][0];
    v16bf pf = load_frag_2x16B(pr + kh * 16, pr + kh * 16 + 8);

    // ---- V tile must be fully staged + visible to all waves ---------------
    if (pk) wait_async_le8();                   // V done (K prefetch may run on)
    else    wait_async_0();
    __syncthreads();

    // ---- O += P * V : 16 c-tiles (from sV) --------------------------------
#pragma unroll
    for (int t = 0; t < 16; ++t) {
      v16bf vf = load_frag_32B(&sV[t * 16 + col][kh * 16]);
      oacc[t] = __builtin_amdgcn_wmma_f32_16x16x32_bf16(false, pf, false, vf,
                                                        (short)0, oacc[t], false, false);
    }

    // ---- close iteration: K prefetch landed, protect sV for next overwrite
    wait_async_0();
    __syncthreads();
  }
  // ---- finalize: O /= l, store [B][N][C] (contig C for proj B-fragments) --
#pragma unroll
  for (int j = 0; j < 8; ++j) {
    const float inv = 1.0f / lrow[j];
    const int n = qrow + j + 8 * kh;
#pragma unroll
    for (int t = 0; t < 16; ++t)
      Ob[((size_t)b * ND + n) * CD + t * 16 + col] = (__bf16)(oacc[t][j] * inv);
  }
}

// ---------------------------------------------------------------------------
// Kernel 5: output projection + bias + residual, fp32 out [B][C][N].
// Software-pipelined like the QKV GEMM.
// ---------------------------------------------------------------------------
__global__ void __launch_bounds__(128) proj_kernel(const __bf16* __restrict__ pw,
                                                   const float* __restrict__ pb,
                                                   const __bf16* __restrict__ AT,
                                                   const float* __restrict__ x,
                                                   float* __restrict__ out) {
  const int b    = blockIdx.z;
  const int o0   = blockIdx.x * 64;
  const int n0   = blockIdx.y * 64;
  const int wv   = threadIdx.x >> 5;
  const int lane = threadIdx.x & 31;
  const int col  = lane & 15;
  const int kh   = lane >> 4;
  const int orow = o0 + wv * 16;

  v8f acc[4];
#pragma unroll
  for (int t = 0; t < 4; ++t) acc[t] = v8f_zero();

  const __bf16* arow  = pw + (size_t)(orow + col) * CD;
  const __bf16* bbase = AT + ((size_t)b * ND + n0) * CD;

  v16bf af = load_frag_2x16B(arow + kh * 16, arow + kh * 16 + 8);
  v16bf bf[4];
#pragma unroll
  for (int t = 0; t < 4; ++t)
    bf[t] = load_frag_32B(bbase + (size_t)(t * 16 + col) * CD + kh * 16);

#pragma unroll
  for (int kc = 0; kc < 8; ++kc) {
    v16bf afn, bfn[4];
    const bool more = (kc < 7);
    if (more) {
      const int c1 = (kc + 1) * 32;
      afn = load_frag_2x16B(arow + c1 + kh * 16, arow + c1 + kh * 16 + 8);
#pragma unroll
      for (int t = 0; t < 4; ++t)
        bfn[t] = load_frag_32B(bbase + (size_t)(t * 16 + col) * CD + c1 + kh * 16);
    }
#pragma unroll
    for (int t = 0; t < 4; ++t)
      acc[t] = __builtin_amdgcn_wmma_f32_16x16x32_bf16(false, af, false, bf[t],
                                                       (short)0, acc[t], false, false);
    if (more) {
      af = afn;
#pragma unroll
      for (int t = 0; t < 4; ++t) bf[t] = bfn[t];
    }
  }
#pragma unroll
  for (int t = 0; t < 4; ++t) {
#pragma unroll
    for (int j = 0; j < 8; ++j) {
      const int o = orow + j + 8 * kh;
      const int n = n0 + t * 16 + col;
      const size_t idx = ((size_t)b * CD + o) * ND + n;
      out[idx] = acc[t][j] + pb[o] + x[idx];
    }
  }
}

// ---------------------------------------------------------------------------
// Launcher. Inputs: x, norm_w, norm_b, qkv_w, qkv_b, proj_w, proj_b (all f32).
// Workspace layout (bytes):
//   xnT  @ 0         : B*N*C bf16   = 4 MiB   [B][N][C]
//   qw   @ 4194304   : 3C*C  bf16   = 384 KiB
//   pw   @ 4587520   : C*C   bf16   = 128 KiB
//   Qb   @ 4718592   : B*N*C bf16   = 4 MiB   [B][N][C]
//   Kb   @ 8912896   : B*N*C bf16   = 4 MiB   [B][N][C]
//   Vb   @ 13107200  : B*C*N bf16   = 4 MiB   [B][C][N]
//   AT   @ 17301504  : B*N*C bf16   = 4 MiB   [B][N][C]
//   total ~20.5 MiB
// ---------------------------------------------------------------------------
extern "C" void kernel_launch(void* const* d_in, const int* in_sizes, int n_in,
                              void* d_out, int out_size, void* d_ws, size_t ws_size,
                              hipStream_t stream) {
  const float* x      = (const float*)d_in[0];
  const float* norm_w = (const float*)d_in[1];
  const float* norm_b = (const float*)d_in[2];
  const float* qkv_w  = (const float*)d_in[3];
  const float* qkv_b  = (const float*)d_in[4];
  const float* proj_w = (const float*)d_in[5];
  const float* proj_b = (const float*)d_in[6];
  float* out = (float*)d_out;

  char* ws = (char*)d_ws;
  __bf16* xnT = (__bf16*)(ws + 0);
  __bf16* qw  = (__bf16*)(ws + 4194304);
  __bf16* pw  = (__bf16*)(ws + 4587520);
  __bf16* Qb  = (__bf16*)(ws + 4718592);
  __bf16* Kb  = (__bf16*)(ws + 8912896);
  __bf16* Vb  = (__bf16*)(ws + 13107200);
  __bf16* AT  = (__bf16*)(ws + 17301504);

  gn_kernel<<<dim3(BD * NGRP), dim3(256), 0, stream>>>(x, norm_w, norm_b, xnT);
  cvt_kernel<<<dim3((3 * CD * CD + 255) / 256), dim3(256), 0, stream>>>(qkv_w, proj_w, qw, pw);
  qkv_kernel<<<dim3(3 * CD / 64, ND / 64, BD), dim3(128), 0, stream>>>(qw, qkv_b, xnT, Qb, Kb, Vb);
  flash_kernel<<<dim3(ND / 64, BD), dim3(128), 0, stream>>>(Qb, Kb, Vb, AT);
  proj_kernel<<<dim3(CD / 64, ND / 64, BD), dim3(128), 0, stream>>>(pw, proj_b, AT, x, out);
}
